// EquivariantPushNet_15092515078738
// MI455X (gfx1250) — matrix operand order repaired and checked by reading, not compiled, not run on hardware
//
#include <hip/hip_runtime.h>

typedef __attribute__((ext_vector_type(16))) _Float16 v16h;
typedef __attribute__((ext_vector_type(8)))  float    v8f;
typedef _Float16 half_t;

#define SIGMA2 0.18f

// ---------------------------------------------------------------------------
// WMMA GEMM helper:  Out[32][N] = A[32][K] @ W[N][K]^T + bias   (optionally ReLU)
// One workgroup of 128 threads (4 waves); each wave owns strided 16x16 tiles.
// Fragment packing follows CDNA5 ISA 16-bit A(16x32)/B(32x16)/D(16x16) layouts.
// ---------------------------------------------------------------------------
template <typename AT, typename OT>
__device__ __forceinline__ void wmma_gemm(const AT* __restrict__ A, int K,
                                          const float* __restrict__ W,
                                          const float* __restrict__ bias,
                                          OT* __restrict__ Out, int N,
                                          bool relu, int tid) {
  const int wave = tid >> 5;
  const int lane = tid & 31;
  const int n16  = lane & 15;   // also the A-row index m
  const int hi   = lane >> 4;
  const int ncols  = N >> 4;
  const int ntiles = 2 * ncols;
  for (int t = wave; t < ntiles; t += 4) {
    const int mr = t / ncols;
    const int nc = t - mr * ncols;
    const float bval = bias[nc * 16 + n16];
    v8f acc;
#pragma unroll
    for (int r = 0; r < 8; ++r) acc[r] = bval;
    const AT*    Arow = A + (size_t)(mr * 16 + n16) * K;
    const float* Wrow = W + (size_t)(nc * 16 + n16) * K;
    for (int kb = 0; kb < K; kb += 32) {
      v16h av, bvv;
#pragma unroll
      for (int i = 0; i < 16; ++i) {
        const int ka = kb + ((i < 8) ? i : i + 8) + hi * 8;   // A 16x32 layout
        const int kk = kb + ((i < 8) ? i : i + 8) + hi * 8;   // B 32x16 layout
        av[i]  = (_Float16)Arow[ka];
        bvv[i] = (_Float16)Wrow[kk];
      }
      acc = __builtin_amdgcn_wmma_f32_16x16x32_f16(false, av, false, bvv,
                                                   (short)0, acc, false, false);
    }
#pragma unroll
    for (int r = 0; r < 8; ++r) {
      float v = acc[r];
      if (relu) v = fmaxf(v, 0.f);
      Out[(size_t)(mr * 16 + r + hi * 8) * N + nc * 16 + n16] = (OT)v;
    }
  }
}

// ---------------------------------------------------------------------------
// Kernel A: nearest-downsample masks (1024->128), mask sum, binary centroid.
// grid = 32 (b*2+n), block = 512
// ---------------------------------------------------------------------------
__global__ __launch_bounds__(512) void k_mask(const float* __restrict__ masks,
                                              float* __restrict__ mds,
                                              float* __restrict__ inv_msum,
                                              float* __restrict__ pos) {
  __shared__ float4 red[512];
  const int bn  = blockIdx.x;
  const int tid = threadIdx.x;
  const float* mp = masks + (size_t)bn * 1024 * 1024;
  float* dp = mds + (size_t)bn * 16384;
  float s = 0.f, cnt = 0.f, sy = 0.f, sx = 0.f;
  for (int p = tid; p < 16384; p += 512) {
    const int h = p >> 7, w = p & 127;
    const float v = mp[(size_t)(h * 8) * 1024 + w * 8];
    dp[p] = v;
    s += v;
    if (v > 0.5f) { cnt += 1.f; sy += (float)h; sx += (float)w; }
  }
  red[tid] = make_float4(s, cnt, sy, sx);
  __syncthreads();
  for (int off = 256; off > 0; off >>= 1) {
    if (tid < off) {
      float4 a = red[tid], b = red[tid + off];
      red[tid] = make_float4(a.x + b.x, a.y + b.y, a.z + b.z, a.w + b.w);
    }
    __syncthreads();
  }
  if (tid == 0) {
    const float4 r = red[0];
    inv_msum[bn] = 1.f / (r.x + 1e-6f);
    const float cm = fmaxf(r.y, 1.f);
    const float cy = r.z / cm, cx = r.w / cm;
    pos[bn * 2 + 0] = cx * (2.f / 128.f) - 1.f;   // pos_x
    pos[bn * 2 + 1] = cy * (2.f / 128.f) - 1.f;   // pos_y
  }
}

// ---------------------------------------------------------------------------
// Kernel B: node pooling  nodes0[b,n,c] = sum_hw x[b,c,hw]*mds[b,n,hw] * inv_msum
// grid = B*C = 2048, block = 256, float4 streaming over x.
// ---------------------------------------------------------------------------
__global__ __launch_bounds__(256) void k_pool(const float* __restrict__ x,
                                              const float* __restrict__ mds,
                                              const float* __restrict__ inv_msum,
                                              float* __restrict__ n0out) {
  __shared__ float2 red[256];
  const int b = blockIdx.x >> 7;
  const int c = blockIdx.x & 127;
  const int tid = threadIdx.x;
  const float4* x4 = (const float4*)x + ((size_t)blockIdx.x << 12);
  const float4* m0 = (const float4*)mds + ((size_t)(b * 2 + 0) << 12);
  const float4* m1 = (const float4*)mds + ((size_t)(b * 2 + 1) << 12);
  float a0 = 0.f, a1 = 0.f;
#pragma unroll 4
  for (int k = 0; k < 16; ++k) {
    const int i = k * 256 + tid;
    const float4 xv = x4[i], q0 = m0[i], q1 = m1[i];
    a0 += xv.x * q0.x + xv.y * q0.y + xv.z * q0.z + xv.w * q0.w;
    a1 += xv.x * q1.x + xv.y * q1.y + xv.z * q1.z + xv.w * q1.w;
  }
  red[tid] = make_float2(a0, a1);
  __syncthreads();
  for (int off = 128; off > 0; off >>= 1) {
    if (tid < off) {
      float2 p = red[tid], q = red[tid + off];
      red[tid] = make_float2(p.x + q.x, p.y + q.y);
    }
    __syncthreads();
  }
  if (tid == 0) {
    n0out[(b * 2 + 0) * 128 + c] = red[0].x * inv_msum[b * 2 + 0];
    n0out[(b * 2 + 1) * 128 + c] = red[0].y * inv_msum[b * 2 + 1];
  }
}

// ---------------------------------------------------------------------------
// Kernel C: distance-biased MHA + FFN on the 32x128 token matrix, all in LDS,
// GEMMs via v_wmma_f32_16x16x32_f16.  grid = 1, block = 128 (4 waves).
// ---------------------------------------------------------------------------
__global__ __launch_bounds__(128) void k_graph(
    const float* __restrict__ gN0, const float* __restrict__ gPos,
    const float* __restrict__ Wq, const float* __restrict__ bq,
    const float* __restrict__ Wk, const float* __restrict__ bk,
    const float* __restrict__ Wv, const float* __restrict__ bvp,
    const float* __restrict__ Wo, const float* __restrict__ bo,
    const float* __restrict__ ln1w, const float* __restrict__ ln1b,
    const float* __restrict__ W1, const float* __restrict__ b1,
    const float* __restrict__ W2, const float* __restrict__ b2,
    const float* __restrict__ ln2w, const float* __restrict__ ln2b,
    float* __restrict__ gN2) {
  __shared__ float  sX[32 * 128];      // nodes0 -> (in place) nodes1
  __shared__ half_t sH[3 * 4096];      // Q | K | V  (E=H1 aliases Q|K)
  __shared__ float  sAttn[256];
  half_t* sQ = sH;
  half_t* sK = sH + 4096;
  half_t* sV = sH + 8192;
  half_t* sE = sH;                     // 32x256 f16, spans Q|K after they die
  const int tid = threadIdx.x;

  for (int i = tid; i < 4096; i += 128) sX[i] = gN0[i];
  __syncthreads();

  wmma_gemm<float, half_t>(sX, 128, Wq, bq,  sQ, 128, false, tid);
  wmma_gemm<float, half_t>(sX, 128, Wk, bk,  sK, 128, false, tid);
  wmma_gemm<float, half_t>(sX, 128, Wv, bvp, sV, 128, false, tid);
  __syncthreads();

  // scores + distance bias + softmax over the 2 tokens (128 rows = b*8+h*2+i)
  {
    const int row = tid;
    const int b = row >> 3, h = (row >> 1) & 3, i = row & 1;
    const half_t* q  = sQ + (b * 2 + i) * 128 + h * 32;
    const half_t* k0 = sK + (b * 2 + 0) * 128 + h * 32;
    const half_t* k1 = sK + (b * 2 + 1) * 128 + h * 32;
    float s0 = 0.f, s1 = 0.f;
#pragma unroll
    for (int d = 0; d < 32; ++d) {
      const float qv = (float)q[d];
      s0 += qv * (float)k0[d];
      s1 += qv * (float)k1[d];
    }
    const float scale = 0.17677669529663687f;   // 1/sqrt(32)
    s0 *= scale; s1 *= scale;
    const float dxp = gPos[b * 4 + 0] - gPos[b * 4 + 2];
    const float dyp = gPos[b * 4 + 1] - gPos[b * 4 + 3];
    const float d2  = dxp * dxp + dyp * dyp;
    const float dist = (d2 > 0.f) ? sqrtf(d2) : 0.f;
    if (i == 0) s1 -= dist; else s0 -= dist;
    const float mx = fmaxf(s0, s1);
    const float e0 = __expf(s0 - mx), e1 = __expf(s1 - mx);
    const float inv = 1.f / (e0 + e1);
    sAttn[row * 2 + 0] = e0 * inv;
    sAttn[row * 2 + 1] = e1 * inv;
  }
  __syncthreads();

  // attn @ V  -> overwrite sQ (Q dead)
  for (int e = tid; e < 4096; e += 128) {
    const int t = e >> 7, c = e & 127;
    const int b = t >> 1, i = t & 1, h = c >> 5;
    const float a0 = sAttn[((b * 4 + h) * 2 + i) * 2 + 0];
    const float a1 = sAttn[((b * 4 + h) * 2 + i) * 2 + 1];
    const float v = a0 * (float)sV[(b * 2 + 0) * 128 + c] +
                    a1 * (float)sV[(b * 2 + 1) * 128 + c];
    sQ[e] = (half_t)v;
  }
  __syncthreads();

  wmma_gemm<half_t, half_t>(sQ, 128, Wo, bo, sK, 128, false, tid);  // out-proj
  __syncthreads();

  // LN1: nodes1 = LN(nodes0 + attn_out), in place into sX
  if (tid < 32) {
    float* xr = sX + tid * 128;
    const half_t* orow = sK + tid * 128;
    float m = 0.f;
    for (int c = 0; c < 128; ++c) m += xr[c] + (float)orow[c];
    m *= (1.f / 128.f);
    float v = 0.f;
    for (int c = 0; c < 128; ++c) {
      const float d = xr[c] + (float)orow[c] - m;
      v += d * d;
    }
    const float rstd = rsqrtf(v * (1.f / 128.f) + 1e-5f);
    for (int c = 0; c < 128; ++c)
      xr[c] = (xr[c] + (float)orow[c] - m) * rstd * ln1w[c] + ln1b[c];
  }
  __syncthreads();

  wmma_gemm<float, half_t>(sX, 128, W1, b1, sE, 256, true, tid);    // FFN up+ReLU
  __syncthreads();
  wmma_gemm<half_t, half_t>(sE, 256, W2, b2, sV, 128, false, tid);  // FFN down
  __syncthreads();

  // LN2: nodes2 = LN(nodes1 + ffn) -> global
  if (tid < 32) {
    const float* xr = sX + tid * 128;
    const half_t* fr = sV + tid * 128;
    float m = 0.f;
    for (int c = 0; c < 128; ++c) m += xr[c] + (float)fr[c];
    m *= (1.f / 128.f);
    float v = 0.f;
    for (int c = 0; c < 128; ++c) {
      const float d = xr[c] + (float)fr[c] - m;
      v += d * d;
    }
    const float rstd = rsqrtf(v * (1.f / 128.f) + 1e-5f);
    for (int c = 0; c < 128; ++c)
      gN2[tid * 128 + c] = (xr[c] + (float)fr[c] - m) * rstd * ln2w[c] + ln2b[c];
  }
}

// ---------------------------------------------------------------------------
// Kernel D: separable Gaussian broadcast + residual.
// grid = B*C = 2048, block = 256, float4 streaming.
// ---------------------------------------------------------------------------
__global__ __launch_bounds__(256) void k_bcast(const float* __restrict__ x,
                                               const float* __restrict__ gN2,
                                               const float* __restrict__ gPos,
                                               float* __restrict__ out) {
  __shared__ float ex0[128], ex1[128], ey0[128], ey1[128];
  const int b = blockIdx.x >> 7;
  const int c = blockIdx.x & 127;
  const int tid = threadIdx.x;
  const float px0 = gPos[b * 4 + 0], py0 = gPos[b * 4 + 1];
  const float px1 = gPos[b * 4 + 2], py1 = gPos[b * 4 + 3];
  if (tid < 128) {
    const float g = 2.f * tid * (1.f / 127.f) - 1.f;
    const float d0 = g - px0, d1 = g - px1;
    ex0[tid] = __expf(-d0 * d0 * (1.f / SIGMA2));
    ex1[tid] = __expf(-d1 * d1 * (1.f / SIGMA2));
  } else {
    const int h = tid - 128;
    const float g = 2.f * h * (1.f / 127.f) - 1.f;
    const float d0 = g - py0, d1 = g - py1;
    ey0[h] = __expf(-d0 * d0 * (1.f / SIGMA2));
    ey1[h] = __expf(-d1 * d1 * (1.f / SIGMA2));
  }
  const float n0 = gN2[(b * 2 + 0) * 128 + c];
  const float n1 = gN2[(b * 2 + 1) * 128 + c];
  __syncthreads();
  const size_t base4 = ((size_t)blockIdx.x) << 12;
  const float4* x4 = (const float4*)x + base4;
  float4* o4 = (float4*)out + base4;
#pragma unroll 4
  for (int k = 0; k < 16; ++k) {
    const int i4 = k * 256 + tid;
    const int h = i4 >> 5;
    const int w = (i4 & 31) << 2;
    const float4 xv = x4[i4];
    const float w0 = n0 * ey0[h], w1 = n1 * ey1[h];
    float4 r;
    r.x = xv.x + w0 * ex0[w + 0] + w1 * ex1[w + 0];
    r.y = xv.y + w0 * ex0[w + 1] + w1 * ex1[w + 1];
    r.z = xv.z + w0 * ex0[w + 2] + w1 * ex1[w + 2];
    r.w = xv.w + w0 * ex0[w + 3] + w1 * ex1[w + 3];
    o4[i4] = r;
  }
}

// ---------------------------------------------------------------------------
extern "C" void kernel_launch(void* const* d_in, const int* in_sizes, int n_in,
                              void* d_out, int out_size, void* d_ws, size_t ws_size,
                              hipStream_t stream) {
  const float* x     = (const float*)d_in[0];
  const float* masks = (const float*)d_in[1];
  const float* Wq = (const float*)d_in[2];  const float* bq = (const float*)d_in[3];
  const float* Wk = (const float*)d_in[4];  const float* bk = (const float*)d_in[5];
  const float* Wv = (const float*)d_in[6];  const float* bv = (const float*)d_in[7];
  const float* Wo = (const float*)d_in[8];  const float* bo = (const float*)d_in[9];
  const float* ln1w = (const float*)d_in[10]; const float* ln1b = (const float*)d_in[11];
  const float* W1 = (const float*)d_in[12]; const float* b1 = (const float*)d_in[13];
  const float* W2 = (const float*)d_in[14]; const float* b2 = (const float*)d_in[15];
  const float* ln2w = (const float*)d_in[16]; const float* ln2b = (const float*)d_in[17];

  float* ws       = (float*)d_ws;
  float* mds      = ws;              // 32 * 16384        (2 MB, L2-resident)
  float* inv_msum = ws + 524288;     // 32
  float* pos      = ws + 524320;     // 64
  float* n0       = ws + 524384;     // 32 * 128
  float* n2       = ws + 528480;     // 32 * 128
  float* out      = (float*)d_out;

  hipLaunchKernelGGL(k_mask,  dim3(32),   dim3(512), 0, stream, masks, mds, inv_msum, pos);
  hipLaunchKernelGGL(k_pool,  dim3(2048), dim3(256), 0, stream, x, mds, inv_msum, n0);
  hipLaunchKernelGGL(k_graph, dim3(1),    dim3(128), 0, stream, n0, pos,
                     Wq, bq, Wk, bk, Wv, bv, Wo, bo,
                     ln1w, ln1b, W1, b1, W2, b2, ln2w, ln2b, n2);
  hipLaunchKernelGGL(k_bcast, dim3(2048), dim3(256), 0, stream, x, n2, pos, out);
}
